// CausalMultiHeadSelfAttention_26439818674390
// MI455X (gfx1250) — matrix operand order repaired and checked by reading, not compiled, not run on hardware
//
#include <hip/hip_runtime.h>

// ---------------------------------------------------------------------------
// Causal MHA + RoPE for MI455X (gfx1250, wave32, WMMA).
// All matmuls in bf16 WMMA (f32 accum); softmax/RoPE in fp32 VALU.
// ---------------------------------------------------------------------------

#define D_MODEL 1024
#define N_HEADS 16
#define D_KV    64
#define SEQ     2048
#define BATCH   2
#define NTOK    (BATCH * SEQ)   // 4096 tokens

typedef __attribute__((ext_vector_type(16))) __bf16 v16bf;
typedef __attribute__((ext_vector_type(8)))  __bf16 v8bf;
typedef __attribute__((ext_vector_type(8)))  float  v8f;

__device__ __forceinline__ v8f wmma_bf16(v16bf a, v16bf b, v8f c) {
  // D = A(16x32 bf16) * B(32x16 bf16) + C(16x16 f32)
  return __builtin_amdgcn_wmma_f32_16x16x32_bf16(
      /*neg_a=*/false, a, /*neg_b=*/false, b,
      /*c_mod=*/(short)0, c, /*reuse_a=*/false, /*reuse_b=*/false);
}

// A-fragment (16x32, row-major source, ld in elements).
// Lane L: row m = L&15, g = L>>4; elements e<8 at k=8g+e, e>=8 at k=16+8g+(e-8).
__device__ __forceinline__ v16bf load_a_frag(const __bf16* tile, int ld) {
  const int lane = threadIdx.x & 31;
  const int m = lane & 15, g = lane >> 4;
  const __bf16* p = tile + m * ld + 8 * g;
  v8bf lo = *(const v8bf*)(p);
  v8bf hi = *(const v8bf*)(p + 16);
  v16bf a;
#pragma unroll
  for (int i = 0; i < 8; ++i) { a[i] = lo[i]; a[i + 8] = hi[i]; }
  return a;
}

// B-fragment (32x16). Source rows are the N-columns (i.e. src[n][k] = B[k][n]),
// ld in elements. Lane L: column n = L&15, g = L>>4; elements e<8 at k=8g+e, e>=8 at k=16+8g+(e-8).
__device__ __forceinline__ v16bf load_b_frag(const __bf16* tile, int ld) {
  const int lane = threadIdx.x & 31;
  const int n = lane & 15, g = lane >> 4;
  const __bf16* p = tile + n * ld + 8 * g;
  v8bf lo = *(const v8bf*)(p);
  v8bf hi = *(const v8bf*)(p + 16);
  v16bf b;
#pragma unroll
  for (int i = 0; i < 8; ++i) { b[i] = lo[i]; b[i + 8] = hi[i]; }
  return b;
}

// ---------------------------------------------------------------------------
// fp32 -> bf16 conversion
// ---------------------------------------------------------------------------
__global__ void cvt_bf16_kernel(const float* __restrict__ src,
                                __bf16* __restrict__ dst, int n) {
  int i = blockIdx.x * blockDim.x + threadIdx.x;
  if (i < n) dst[i] = (__bf16)src[i];
}

// ---------------------------------------------------------------------------
// RoPE tables: cos/sin[pos][i], i = 0..31, freq = theta^(-2i/64)
// ---------------------------------------------------------------------------
__global__ void rope_tables_kernel(float* __restrict__ cosT,
                                   float* __restrict__ sinT) {
  int i = blockIdx.x * blockDim.x + threadIdx.x;
  if (i < SEQ * 32) {
    int pos = i >> 5, f = i & 31;
    float freq = powf(10000.0f, -(2.0f * (float)f) / 64.0f);
    float ang = (float)pos * freq;
    cosT[i] = cosf(ang);
    sinT[i] = sinf(ang);
  }
}

// ---------------------------------------------------------------------------
// Q/K projection: Out = X(4096x1024) * W^T, + RoPE, store bf16 [b,h,s,d].
// One wave: 16 (tokens) x 64 (features). 4096 waves -> 512 blocks x 8 waves.
// ---------------------------------------------------------------------------
__global__ __launch_bounds__(256) void qk_proj_rope_kernel(
    const __bf16* __restrict__ X, const __bf16* __restrict__ W,
    const int* __restrict__ tpos, const float* __restrict__ cosT,
    const float* __restrict__ sinT, __bf16* __restrict__ Out) {
  const int gw = blockIdx.x * 8 + (threadIdx.x >> 5);
  const int m0 = (gw >> 4) * 16;   // token tile
  const int n0 = (gw & 15) * 64;   // feature group
  const int lane = threadIdx.x & 31;
  const int nn = lane & 15, g = lane >> 4;

  v8f acc[4] = {};
  for (int k0 = 0; k0 < D_MODEL; k0 += 32) {
    v16bf a = load_a_frag(X + (size_t)m0 * D_MODEL + k0, D_MODEL);
#pragma unroll
    for (int t = 0; t < 4; ++t)
      acc[t] = wmma_bf16(
          a, load_b_frag(W + (size_t)(n0 + t * 16) * D_MODEL + k0, D_MODEL),
          acc[t]);
  }

#pragma unroll
  for (int t = 0; t < 4; ++t) {
    const int col = n0 + t * 16 + nn;        // output feature
    const int h = col >> 6, d = col & 63;
#pragma unroll
    for (int v = 0; v < 8; ++v) {
      const int row = m0 + v + 8 * g;        // global token
      const int bb = row >> 11, s = row & (SEQ - 1);
      const int pos = tpos[s];
      const float cv = cosT[pos * 32 + (d >> 1)];
      const float sv = sinT[pos * 32 + (d >> 1)];
      const float val = acc[t][v];
      const float part = __shfl_xor(val, 1, 32);  // pair partner (even<->odd d)
      const float r = (d & 1) ? (cv * val + sv * part)
                              : (cv * val - sv * part);
      Out[(((size_t)(bb * N_HEADS + h) * SEQ + s) << 6) + d] = (__bf16)r;
    }
  }
}

// ---------------------------------------------------------------------------
// V projection, transposed: C = Wv(1024x1024) * X^T -> Vt bf16 [b,h,d,s].
// One wave: 16 (features) x 64 (tokens). 64*64 = 4096 waves.
// ---------------------------------------------------------------------------
__global__ __launch_bounds__(256) void v_proj_kernel(
    const __bf16* __restrict__ Wv, const __bf16* __restrict__ X,
    __bf16* __restrict__ Vt) {
  const int gw = blockIdx.x * 8 + (threadIdx.x >> 5);
  const int m0 = (gw >> 6) * 16;   // feature tile (0..63)
  const int n0 = (gw & 63) * 64;   // token group  (0..63)
  const int lane = threadIdx.x & 31;
  const int nn = lane & 15, g = lane >> 4;

  v8f acc[4] = {};
  for (int k0 = 0; k0 < D_MODEL; k0 += 32) {
    v16bf a = load_a_frag(Wv + (size_t)m0 * D_MODEL + k0, D_MODEL);
#pragma unroll
    for (int t = 0; t < 4; ++t)
      acc[t] = wmma_bf16(
          a, load_b_frag(X + (size_t)(n0 + t * 16) * D_MODEL + k0, D_MODEL),
          acc[t]);
  }

#pragma unroll
  for (int t = 0; t < 4; ++t) {
#pragma unroll
    for (int v = 0; v < 8; ++v) {
      const int feat = m0 + v + 8 * g;
      const int h = feat >> 6, d = feat & 63;
      const int tok = n0 + t * 16 + nn;
      const int bb = tok >> 11, s = tok & (SEQ - 1);
      Vt[(size_t)(((bb * N_HEADS + h) << 6) + d) * SEQ + s] = (__bf16)acc[t][v];
    }
  }
}

// ---------------------------------------------------------------------------
// Flash attention: one wave per 16 q-rows; K-tiles of 32 with online softmax.
// Q,K: bf16 [b,h,s,d]; Vt: bf16 [b,h,d,s]; out: bf16 [token, h*64+d].
// ---------------------------------------------------------------------------
__global__ __launch_bounds__(256) void flash_attn_kernel(
    const __bf16* __restrict__ Q, const __bf16* __restrict__ K,
    const __bf16* __restrict__ Vt, __bf16* __restrict__ attnOut) {
  const int bh = blockIdx.x >> 4;                 // 0..31
  const int q0 = ((blockIdx.x & 15) * 8 + (threadIdx.x >> 5)) * 16;
  const int lane = threadIdx.x & 31;
  const int nn = lane & 15, g = lane >> 4;
  const int bb = bh >> 4, h = bh & 15;

  const __bf16* Qb = Q + (size_t)bh * SEQ * D_KV;
  const __bf16* Kb = K + (size_t)bh * SEQ * D_KV;
  const __bf16* Vb = Vt + (size_t)bh * D_KV * SEQ;

  __shared__ __attribute__((aligned(64))) __bf16 Pbuf[8 * 16 * 32];
  __bf16* Pw = &Pbuf[(threadIdx.x >> 5) * 512];   // per-wave 16x32 bf16 tile

  const v16bf aq0 = load_a_frag(Qb + (size_t)q0 * D_KV + 0, D_KV);
  const v16bf aq1 = load_a_frag(Qb + (size_t)q0 * D_KV + 32, D_KV);

  v8f o0 = {}, o1 = {}, o2 = {}, o3 = {};
  float mrow[8], lrow[8];
#pragma unroll
  for (int v = 0; v < 8; ++v) { mrow[v] = -3.0e38f; lrow[v] = 0.0f; }

  const int numKt = (q0 + 47) >> 5;   // ceil((q0+16)/32) causal tiles
  for (int kt = 0; kt < numKt; ++kt) {
    const int kt0 = kt * 32;
    if (kt + 1 < numKt)
      __builtin_prefetch(Kb + (size_t)(kt0 + 32) * D_KV, 0, 1);

    v16bf bk00 = load_b_frag(Kb + (size_t)kt0 * D_KV + 0, D_KV);
    v16bf bk01 = load_b_frag(Kb + (size_t)kt0 * D_KV + 32, D_KV);
    v16bf bk10 = load_b_frag(Kb + (size_t)(kt0 + 16) * D_KV + 0, D_KV);
    v16bf bk11 = load_b_frag(Kb + (size_t)(kt0 + 16) * D_KV + 32, D_KV);

    v8f s0 = {}, s1 = {};
    s0 = wmma_bf16(aq0, bk00, s0);
    s0 = wmma_bf16(aq1, bk01, s0);
    s1 = wmma_bf16(aq0, bk10, s1);
    s1 = wmma_bf16(aq1, bk11, s1);

#pragma unroll
    for (int v = 0; v < 8; ++v) {
      const int q = q0 + v + 8 * g;
      float x0 = s0[v] * 0.125f;                  // 1/sqrt(64)
      float x1 = s1[v] * 0.125f;
      if (kt0 + nn > q)      x0 = -3.0e38f;       // causal mask
      if (kt0 + 16 + nn > q) x1 = -3.0e38f;

      float mt = fmaxf(x0, x1);
#pragma unroll
      for (int off = 8; off >= 1; off >>= 1)      // reduce within 16-lane half
        mt = fmaxf(mt, __shfl_xor(mt, off, 32));
      const float nm = fmaxf(mrow[v], mt);
      const float alpha = __expf(mrow[v] - nm);
      mrow[v] = nm;
      const float p0 = __expf(x0 - nm);
      const float p1 = __expf(x1 - nm);
      float rs = p0 + p1;
#pragma unroll
      for (int off = 8; off >= 1; off >>= 1)
        rs += __shfl_xor(rs, off, 32);
      lrow[v] = lrow[v] * alpha + rs;
      o0[v] *= alpha; o1[v] *= alpha; o2[v] *= alpha; o3[v] *= alpha;
      // stage P (C layout -> row-major 16x32) for A-fragment reload
      Pw[(v + 8 * g) * 32 + nn]      = (__bf16)p0;
      Pw[(v + 8 * g) * 32 + 16 + nn] = (__bf16)p1;
    }

    const v16bf ap = load_a_frag(Pw, 32);   // DS ops are in-order per wave
    o0 = wmma_bf16(ap, load_b_frag(Vb + (size_t)0  * SEQ + kt0, SEQ), o0);
    o1 = wmma_bf16(ap, load_b_frag(Vb + (size_t)16 * SEQ + kt0, SEQ), o1);
    o2 = wmma_bf16(ap, load_b_frag(Vb + (size_t)32 * SEQ + kt0, SEQ), o2);
    o3 = wmma_bf16(ap, load_b_frag(Vb + (size_t)48 * SEQ + kt0, SEQ), o3);
  }

#pragma unroll
  for (int v = 0; v < 8; ++v) {
    const float inv = 1.0f / lrow[v];
    const int tok = bb * SEQ + q0 + v + 8 * g;
    __bf16* dst = attnOut + (size_t)tok * D_MODEL + h * D_KV + nn;
    dst[0]  = (__bf16)(o0[v] * inv);
    dst[16] = (__bf16)(o1[v] * inv);
    dst[32] = (__bf16)(o2[v] * inv);
    dst[48] = (__bf16)(o3[v] * inv);
  }
}

// ---------------------------------------------------------------------------
// Output projection: out(fp32) = attn(4096x1024 bf16) * Wo^T
// ---------------------------------------------------------------------------
__global__ __launch_bounds__(256) void o_proj_kernel(
    const __bf16* __restrict__ A, const __bf16* __restrict__ W,
    float* __restrict__ Outf) {
  const int gw = blockIdx.x * 8 + (threadIdx.x >> 5);
  const int m0 = (gw >> 4) * 16;
  const int n0 = (gw & 15) * 64;
  const int lane = threadIdx.x & 31;
  const int nn = lane & 15, g = lane >> 4;

  v8f acc[4] = {};
  for (int k0 = 0; k0 < D_MODEL; k0 += 32) {
    v16bf a = load_a_frag(A + (size_t)m0 * D_MODEL + k0, D_MODEL);
#pragma unroll
    for (int t = 0; t < 4; ++t)
      acc[t] = wmma_bf16(
          a, load_b_frag(W + (size_t)(n0 + t * 16) * D_MODEL + k0, D_MODEL),
          acc[t]);
  }
#pragma unroll
  for (int t = 0; t < 4; ++t)
#pragma unroll
    for (int v = 0; v < 8; ++v)
      Outf[(size_t)(m0 + v + 8 * g) * D_MODEL + n0 + t * 16 + nn] = acc[t][v];
}

// ---------------------------------------------------------------------------
// Host launcher. Workspace layout (all 256B-aligned, ~48.6 MB total):
//   x_bf16, wq/wk/wv/wo bf16, Q, K, Vt, attn, cosT, sinT
// ---------------------------------------------------------------------------
static inline size_t align256(size_t x) { return (x + 255) & ~(size_t)255; }

extern "C" void kernel_launch(void* const* d_in, const int* in_sizes, int n_in,
                              void* d_out, int out_size, void* d_ws,
                              size_t ws_size, hipStream_t stream) {
  const float* x  = (const float*)d_in[0];
  const float* wq = (const float*)d_in[1];
  const float* wk = (const float*)d_in[2];
  const float* wv = (const float*)d_in[3];
  const float* wo = (const float*)d_in[4];
  const int* tpos = (const int*)d_in[5];

  char* p = (char*)d_ws;
  auto alloc = [&](size_t bytes) { char* r = p; p += align256(bytes); return r; };

  __bf16* xb   = (__bf16*)alloc((size_t)NTOK * D_MODEL * 2);
  __bf16* wqb  = (__bf16*)alloc((size_t)D_MODEL * D_MODEL * 2);
  __bf16* wkb  = (__bf16*)alloc((size_t)D_MODEL * D_MODEL * 2);
  __bf16* wvb  = (__bf16*)alloc((size_t)D_MODEL * D_MODEL * 2);
  __bf16* wob  = (__bf16*)alloc((size_t)D_MODEL * D_MODEL * 2);
  __bf16* Qb   = (__bf16*)alloc((size_t)NTOK * D_MODEL * 2);
  __bf16* Kb   = (__bf16*)alloc((size_t)NTOK * D_MODEL * 2);
  __bf16* Vtb  = (__bf16*)alloc((size_t)NTOK * D_MODEL * 2);
  __bf16* attn = (__bf16*)alloc((size_t)NTOK * D_MODEL * 2);
  float*  cosT = (float*)alloc((size_t)SEQ * 32 * 4);
  float*  sinT = (float*)alloc((size_t)SEQ * 32 * 4);

  const int nx = NTOK * D_MODEL;       // 4194304
  const int nw = D_MODEL * D_MODEL;    // 1048576
  cvt_bf16_kernel<<<(nx + 255) / 256, 256, 0, stream>>>(x, xb, nx);
  cvt_bf16_kernel<<<(nw + 255) / 256, 256, 0, stream>>>(wq, wqb, nw);
  cvt_bf16_kernel<<<(nw + 255) / 256, 256, 0, stream>>>(wk, wkb, nw);
  cvt_bf16_kernel<<<(nw + 255) / 256, 256, 0, stream>>>(wv, wvb, nw);
  cvt_bf16_kernel<<<(nw + 255) / 256, 256, 0, stream>>>(wo, wob, nw);
  rope_tables_kernel<<<(SEQ * 32 + 255) / 256, 256, 0, stream>>>(cosT, sinT);

  qk_proj_rope_kernel<<<512, 256, 0, stream>>>(xb, wqb, tpos, cosT, sinT, Qb);
  qk_proj_rope_kernel<<<512, 256, 0, stream>>>(xb, wkb, tpos, cosT, sinT, Kb);
  v_proj_kernel<<<512, 256, 0, stream>>>(wvb, xb, Vtb);
  flash_attn_kernel<<<512, 256, 0, stream>>>(Qb, Kb, Vtb, attn);
  o_proj_kernel<<<512, 256, 0, stream>>>(attn, wob, (float*)d_out);
}
